// GCNConv_74131135529465
// MI455X (gfx1250) — matrix operand order repaired and checked by reading, not compiled, not run on hardware
//
#include <hip/hip_runtime.h>
#include <hip/hip_bf16.h>
#include <stdint.h>

#define D 512
#define EPSV 1e-5f

typedef __attribute__((ext_vector_type(16))) __bf16       v16bf;
typedef __attribute__((ext_vector_type(8)))  float        v8f;
typedef __attribute__((ext_vector_type(4))) unsigned int  u32x4;
typedef __attribute__((ext_vector_type(4)))  float        f32x4;

__device__ __forceinline__ unsigned short f2bf(float f) {
  unsigned int u = __float_as_uint(f);
  u += 0x7FFFu + ((u >> 16) & 1u);   // round-to-nearest-even
  return (unsigned short)(u >> 16);
}

// CDNA5 async memory->LDS copy (ASYNCcnt-tracked). VDST = 32-bit LDS byte
// address (low 32 bits of the generic pointer), VADDR = 64-bit global address.
__device__ __forceinline__ void async_copy_b128(unsigned lds_addr, const void* gaddr) {
  asm volatile("global_load_async_to_lds_b128 %0, %1, off"
               :: "v"(lds_addr), "v"(gaddr) : "memory");
}
__device__ __forceinline__ void wait_async0() {
  asm volatile("s_wait_asynccnt 0x0" ::: "memory");
}

// ---------------- utility kernels ----------------

__global__ void k_zero4(f32x4* p, int n4) {
  int i = blockIdx.x * 256 + threadIdx.x;
  if (i < n4) { f32x4 z = {0.f, 0.f, 0.f, 0.f}; p[i] = z; }
}

__global__ void k_deg(const int* __restrict__ src, const int* __restrict__ dst,
                      float* outdeg, float* indeg, int E) {
  int i = blockIdx.x * 256 + threadIdx.x;
  if (i < E) {
    atomicAdd(&outdeg[src[i]], 1.0f);
    atomicAdd(&indeg[dst[i]], 1.0f);
  }
}

__global__ void k_degnorm(float* a, float* b, int N) {
  int i = blockIdx.x * 256 + threadIdx.x;
  if (i < N) {
    a[i] = rsqrtf(fmaxf(a[i], 1.0f));
    b[i] = rsqrtf(fmaxf(b[i], 1.0f));
  }
}

// Wt[l][n][k] = bf16(W[l][k][n])
__global__ void k_transW(const float* __restrict__ W, unsigned short* __restrict__ Wt, int total) {
  int i = blockIdx.x * 256 + threadIdx.x;
  if (i >= total) return;
  int l = i >> 18;            // D*D = 262144
  int rem = i & 262143;
  int n = rem >> 9;
  int k = rem & 511;
  Wt[i] = f2bf(W[(size_t)l * D * D + (size_t)k * D + n]);
}

// ---------------- batchnorm reductions ----------------

__global__ void k_bn_partial(const float* __restrict__ x, float* psum, float* psq,
                             int N, int rps) {
  int c  = threadIdx.x & 63;
  int rt = threadIdx.x >> 6;
  int col = blockIdx.x * 64 + c;
  int r    = blockIdx.y * rps + rt;
  int rend = min(blockIdx.y * rps + rps, N);
  float s = 0.f, q = 0.f;
  for (; r < rend; r += 4) {
    float v = x[(size_t)r * D + col];
    s += v; q += v * v;
  }
  __shared__ float sh[2][4][64];
  sh[0][rt][c] = s; sh[1][rt][c] = q;
  __syncthreads();
  if (rt == 0) {
    s = sh[0][0][c] + sh[0][1][c] + sh[0][2][c] + sh[0][3][c];
    q = sh[1][0][c] + sh[1][1][c] + sh[1][2][c] + sh[1][3][c];
    psum[blockIdx.y * D + col] = s;
    psq [blockIdx.y * D + col] = q;
  }
}

__global__ void k_bn_final(const float* __restrict__ psum, const float* __restrict__ psq,
                           float* mu, float* rstd, int N) {
  int col = blockIdx.x * 256 + threadIdx.x;
  if (col >= D) return;
  float s = 0.f, q = 0.f;
  for (int i = 0; i < 64; ++i) { s += psum[i * D + col]; q += psq[i * D + col]; }
  float m = s / (float)N;
  float var = q / (float)N - m * m;
  mu[col] = m;
  rstd[col] = rsqrtf(fmaxf(var, 0.f) + EPSV);
}

// h = ((x - mu) * rstd * gamma + beta) * src_norm[row]
__global__ void k_norm(const float* __restrict__ x, const float* __restrict__ mu,
                       const float* __restrict__ rstd, const float* __restrict__ g,
                       const float* __restrict__ be, const float* __restrict__ srcnorm,
                       float* __restrict__ h, int N) {
  int i = blockIdx.x * 256 + threadIdx.x;
  int total = N * (D / 4);
  if (i >= total) return;
  int row = i >> 7;
  int c4  = i & 127;
  f32x4 xv = *(const f32x4*)(x + (size_t)row * D + c4 * 4);
  f32x4 m  = ((const f32x4*)mu)[c4];
  f32x4 rs = ((const f32x4*)rstd)[c4];
  f32x4 gv = ((const f32x4*)g)[c4];
  f32x4 bv = ((const f32x4*)be)[c4];
  float sn = srcnorm[row];
  f32x4 o = ((xv - m) * rs * gv + bv) * sn;
  *(f32x4*)(h + (size_t)row * D + c4 * 4) = o;
}

// agg[dst] += h[src] (vectorized gather + 4 scalar f32 atomics)
__global__ void k_scatter(const float* __restrict__ h, const int* __restrict__ src,
                          const int* __restrict__ dst, float* agg, long long total) {
  long long i = (long long)blockIdx.x * 256 + threadIdx.x;
  if (i >= total) return;
  int e  = (int)(i >> 7);
  int c4 = (int)(i & 127);
  int s = src[e], t = dst[e];
  f32x4 v = *(const f32x4*)(h + (size_t)s * D + c4 * 4);
  float* p = agg + (size_t)t * D + c4 * 4;
  atomicAdd(p + 0, v.x); atomicAdd(p + 1, v.y);
  atomicAdd(p + 2, v.z); atomicAdd(p + 3, v.w);
}

// A_bf16[row][k] = bf16(agg[row][k] * dst_norm[row]); zero pad rows >= N
__global__ void k_scale_cvt(const float* __restrict__ agg, const float* __restrict__ dstnorm,
                            unsigned short* __restrict__ A, int N, int Mpad) {
  int i = blockIdx.x * 256 + threadIdx.x;
  int total = Mpad * (D / 8);
  if (i >= total) return;
  int row = i >> 6;
  int c8  = (i & 63) * 8;
  u32x4 out = {0u, 0u, 0u, 0u};
  if (row < N) {
    float dn = dstnorm[row];
    const float* p = agg + (size_t)row * D + c8;
    f32x4 a = *(const f32x4*)p       * dn;
    f32x4 b = *(const f32x4*)(p + 4) * dn;
    out.x = (unsigned)f2bf(a.x) | ((unsigned)f2bf(a.y) << 16);
    out.y = (unsigned)f2bf(a.z) | ((unsigned)f2bf(a.w) << 16);
    out.z = (unsigned)f2bf(b.x) | ((unsigned)f2bf(b.y) << 16);
    out.w = (unsigned)f2bf(b.z) | ((unsigned)f2bf(b.w) << 16);
  }
  *(u32x4*)(A + (size_t)row * D + c8) = out;
}

// ---------------- WMMA GEMM: out = relu(A_bf16 @ Wt^T + bias) ----------------
// Block: 256 thr = 8 waves. Block tile 128 rows x 64 cols; wave tile 16 x 64.
// B chunks staged LDS-direct via global_load_async_to_lds_b128, double-buffered.
__global__ __launch_bounds__(256) void k_gemm(const unsigned short* __restrict__ A,
                                              const unsigned short* __restrict__ Wt,
                                              const float* __restrict__ bias,
                                              float* __restrict__ out, int Mreal) {
  __shared__ __align__(16) unsigned short ldsB[2][64 * 40];  // stride 40: conflict-free b128

  const int lane    = threadIdx.x & 31;
  const int wave    = threadIdx.x >> 5;
  const int nbase   = blockIdx.x * 64;
  const int rowBase = blockIdx.y * 128 + wave * 16;
  const int laneRow = lane & 15;
  const int hi      = lane >> 4;          // 0: lanes 0-15, 1: lanes 16-31

  const unsigned short* aRow = A + (size_t)(rowBase + laneRow) * D;
  const int koffA = hi * 8;               // A frag: lo lanes K0-7/16-23, hi lanes K8-15/24-31
  const int koffB = hi * 8;               // B frag: lo lanes K0-7/16-23, hi lanes K8-15/24-31

  // per-thread staging slot: 16B of the 64x32 bf16 chunk
  const int sn = threadIdx.x >> 2;          // 0..63  (B row = output column)
  const int sk = (threadIdx.x & 3) * 8;     // 0,8,16,24
  const unsigned short* wtp = Wt + (size_t)(nbase + sn) * D + sk;
  const unsigned ldsDst[2] = {
      (unsigned)(uintptr_t)&ldsB[0][sn * 40 + sk],
      (unsigned)(uintptr_t)&ldsB[1][sn * 40 + sk]};

  v8f z8 = {0.f, 0.f, 0.f, 0.f, 0.f, 0.f, 0.f, 0.f};
  v8f acc[4] = {z8, z8, z8, z8};

  // preload k-chunk 0 into buffer 0 (DMA, no VGPR round-trip)
  async_copy_b128(ldsDst[0], wtp);

  int cur = 0;
  for (int k0 = 0; k0 < D; k0 += 32) {
    wait_async0();        // this wave's staged bytes are in LDS
    __syncthreads();      // everyone's are

    if (k0 + 32 < D)      // kick DMA for next chunk into the other buffer
      async_copy_b128(ldsDst[cur ^ 1], wtp + k0 + 32);

    if (k0 + 32 < D) __builtin_prefetch(aRow + k0 + 32, 0, 0);  // global_prefetch_b8

    v16bf a;
    *(u32x4*)&a       = *(const u32x4*)(aRow + k0 + koffA);
    *((u32x4*)&a + 1) = *(const u32x4*)(aRow + k0 + koffA + 16);

    const unsigned short* bbase = &ldsB[cur][0];
#pragma unroll
    for (int t = 0; t < 4; ++t) {
      v16bf b;
      const unsigned short* bp = bbase + (t * 16 + laneRow) * 40 + koffB;
      *(u32x4*)&b       = *(const u32x4*)bp;
      *((u32x4*)&b + 1) = *(const u32x4*)(bp + 16);
      acc[t] = __builtin_amdgcn_wmma_f32_16x16x32_bf16(
          false, a, false, b, (short)0, acc[t], false, false);
    }
    cur ^= 1;
  }

  // epilogue: C layout -> VGPR r: rows r (lanes 0-15) / r+8 (lanes 16-31), col = lane&15
#pragma unroll
  for (int t = 0; t < 4; ++t) {
    int col  = nbase + t * 16 + laneRow;
    float bv = bias[col];
#pragma unroll
    for (int r = 0; r < 8; ++r) {
      int row = rowBase + r + hi * 8;
      if (row < Mreal)
        out[(size_t)row * D + col] = fmaxf(acc[t][r] + bv, 0.0f);
    }
  }
}

// ---------------- host launcher ----------------

extern "C" void kernel_launch(void* const* d_in, const int* in_sizes, int n_in,
                              void* d_out, int out_size, void* d_ws, size_t ws_size,
                              hipStream_t stream) {
  const float* x     = (const float*)d_in[0];
  const int*   src   = (const int*)d_in[1];
  const int*   dst   = (const int*)d_in[2];
  const float* gamma = (const float*)d_in[3];
  const float* beta  = (const float*)d_in[4];
  const float* W     = (const float*)d_in[5];
  const float* bias  = (const float*)d_in[6];
  float*       out   = (float*)d_out;

  const int N = in_sizes[0] / D;
  const int E = in_sizes[1];
  const int L = in_sizes[3] / D;
  const int Mpad = ((N + 127) / 128) * 128;

  char* w = (char*)d_ws;
  auto aup = [](size_t v) { return (v + 255) & ~(size_t)255; };
  size_t o = 0;
  float* outdeg = (float*)(w + o);          o += aup((size_t)N * 4);
  float* indeg  = (float*)(w + o);          o += aup((size_t)N * 4);
  float* psum   = (float*)(w + o);          o += aup((size_t)64 * D * 4);
  float* psq    = (float*)(w + o);          o += aup((size_t)64 * D * 4);
  float* mu     = (float*)(w + o);          o += aup((size_t)D * 4);
  float* rstd   = (float*)(w + o);          o += aup((size_t)D * 4);
  float* h      = (float*)(w + o);          o += aup((size_t)N * D * 4);
  float* agg    = (float*)(w + o);          o += aup((size_t)N * D * 4);
  unsigned short* Abf = (unsigned short*)(w + o); o += aup((size_t)Mpad * D * 2);
  unsigned short* Wt  = (unsigned short*)(w + o); o += aup((size_t)L * D * D * 2);
  float* xbuf   = (float*)(w + o);          o += aup((size_t)N * D * 4);
  if (o > ws_size) return;  // workspace too small; bail deterministically

  // --- degrees -> norms (once) ---
  k_zero4<<<(N / 4 + 255) / 256, 256, 0, stream>>>((f32x4*)outdeg, N / 4);
  k_zero4<<<(N / 4 + 255) / 256, 256, 0, stream>>>((f32x4*)indeg, N / 4);
  k_deg<<<(E + 255) / 256, 256, 0, stream>>>(src, dst, outdeg, indeg, E);
  k_degnorm<<<(N + 255) / 256, 256, 0, stream>>>(outdeg, indeg, N);

  // --- transpose + bf16-convert weights (once) ---
  int tw = L * D * D;
  k_transW<<<(tw + 255) / 256, 256, 0, stream>>>(W, Wt, tw);

  const int rps = (N + 63) / 64;
  for (int l = 0; l < L; ++l) {
    const float* xin = (l == 0) ? x : xbuf;
    float* xout = (l == L - 1) ? out : xbuf;

    k_bn_partial<<<dim3(D / 64, 64), 256, 0, stream>>>(xin, psum, psq, N, rps);
    k_bn_final<<<(D + 255) / 256, 256, 0, stream>>>(psum, psq, mu, rstd, N);
    k_norm<<<(N * (D / 4) + 255) / 256, 256, 0, stream>>>(
        xin, mu, rstd, gamma + (size_t)l * D, beta + (size_t)l * D, outdeg, h, N);
    k_zero4<<<(N * (D / 4) + 255) / 256, 256, 0, stream>>>((f32x4*)agg, N * (D / 4));
    long long stot = (long long)E * (D / 4);
    k_scatter<<<(unsigned)((stot + 255) / 256), 256, 0, stream>>>(h, src, dst, agg, stot);
    k_scale_cvt<<<(Mpad * (D / 8) + 255) / 256, 256, 0, stream>>>(agg, indeg, Abf, N, Mpad);
    k_gemm<<<dim3(D / 64, Mpad / 128), 256, 0, stream>>>(
        Abf, Wt + (size_t)l * D * D, bias + (size_t)l * D, xout, N);
  }
}
